// reactionMPNN_74826920231048
// MI455X (gfx1250) — compile-verified, matrix operand order checked by reading, not run on hardware
//
#include <hip/hip_runtime.h>

// ---------------- problem constants ----------------
#define HPAD   320          // H=300 padded to 20 WMMA tiles of 16
#define HREAL  300
#define NNODES 122880       // multiple of 128
#define NEDGES 245760       // multiple of 128
#define NB     4096
#define GDEPTH 3

// ---------------- WMMA types ----------------
typedef __attribute__((ext_vector_type(16))) __bf16 v16bf;
typedef __attribute__((ext_vector_type(8)))  float  v8f;

union Frag {
  unsigned short u[16];
  uint4 q[2];
  v16bf v;
};

__device__ __forceinline__ unsigned short f2bf(float f) {
  unsigned int x = __builtin_bit_cast(unsigned int, f);
  x += 0x7FFFu + ((x >> 16) & 1u);          // round-to-nearest-even
  return (unsigned short)(x >> 16);
}
__device__ __forceinline__ float bf2f(unsigned short u) {
  unsigned int x = ((unsigned int)u) << 16;
  return __builtin_bit_cast(float, x);
}

__device__ __forceinline__ v8f wmma_bf16(const v16bf& a, const v16bf& b, const v8f& c) {
  // D = A(16x32 bf16) * B(32x16 bf16) + C(f32)
  return __builtin_amdgcn_wmma_f32_16x16x32_bf16(false, a, false, b, (short)0, c, false, false);
}

// Shared epilogue: 4 tiles (2M x 2N), bias add, optional relu, bf16 store.
__device__ __forceinline__ void store_tiles(unsigned short* __restrict__ C, size_t row0, int col0,
                                            int lane, const v8f& a00, const v8f& a01,
                                            const v8f& a10, const v8f& a11,
                                            const float* __restrict__ bias, int relu) {
  int n = lane & 15;
  int moff = (lane & 16) ? 8 : 0;
  float bv0 = bias[col0 + n];
  float bv1 = bias[col0 + 16 + n];
#pragma unroll
  for (int r = 0; r < 8; ++r) {
    size_t r0 = row0 + moff + r;
    size_t r1 = r0 + 16;
    float v;
    v = a00[r] + bv0; if (relu) v = fmaxf(v, 0.f); C[r0 * HPAD + col0 + n]      = f2bf(v);
    v = a01[r] + bv1; if (relu) v = fmaxf(v, 0.f); C[r0 * HPAD + col0 + 16 + n] = f2bf(v);
    v = a10[r] + bv0; if (relu) v = fmaxf(v, 0.f); C[r1 * HPAD + col0 + n]      = f2bf(v);
    v = a11[r] + bv1; if (relu) v = fmaxf(v, 0.f); C[r1 * HPAD + col0 + 16 + n] = f2bf(v);
  }
}

// ---------------- weight prep: pad + transpose + bf16 ----------------
__global__ __launch_bounds__(256) void k_prep(
    const float* __restrict__ Wn, const float* __restrict__ bn,
    const float* __restrict__ We, const float* __restrict__ be,
    const float* __restrict__ W1, const float* __restrict__ b1,
    const float* __restrict__ W2, const float* __restrict__ b2,
    unsigned short* __restrict__ Wnt, unsigned short* __restrict__ Wet,
    unsigned short* __restrict__ W1t, unsigned short* __restrict__ W2t,
    float* __restrict__ bnp, float* __restrict__ bep,
    float* __restrict__ b1p, float* __restrict__ b2p) {
  int idx = blockIdx.x * 256 + threadIdx.x;
  if (idx < 3 * HPAD * HPAD) {            // W1t/W2t: [l][c][k] = W[l][k][c]
    int l = idx / (HPAD * HPAD);
    int rem = idx - l * HPAD * HPAD;
    int c = rem / HPAD;
    int k = rem - c * HPAD;
    bool ok = (c < HREAL) && (k < HREAL);
    size_t s = (size_t)l * HREAL * HREAL + (size_t)k * HREAL + c;
    W1t[idx] = f2bf(ok ? W1[s] : 0.f);
    W2t[idx] = f2bf(ok ? W2[s] : 0.f);
  }
  if (idx < HPAD * 64) {                  // Wnt: [c][k], K=64
    int c = idx >> 6, k = idx & 63;
    Wnt[idx] = f2bf((c < HREAL) ? Wn[k * HREAL + c] : 0.f);
  }
  if (idx < HPAD * 32) {                  // Wet: [c][k], K padded 16->32
    int c = idx >> 5, k = idx & 31;
    Wet[idx] = f2bf((c < HREAL && k < 16) ? We[k * HREAL + c] : 0.f);
  }
  if (idx < HPAD) {
    bnp[idx] = (idx < HREAL) ? bn[idx] : 0.f;
    bep[idx] = (idx < HREAL) ? be[idx] : 0.f;
#pragma unroll
    for (int l = 0; l < GDEPTH; ++l) {
      b1p[l * HPAD + idx] = (idx < HREAL) ? b1[l * HREAL + idx] : 0.f;
      b2p[l * HPAD + idx] = (idx < HREAL) ? b2[l * HREAL + idx] : 0.f;
    }
  }
}

// ---------------- main 320-K GEMM: C = act(A @ W + b), bf16 in/out ----------------
__global__ __launch_bounds__(256) void k_gemm320(
    const unsigned short* __restrict__ A, const unsigned short* __restrict__ Wt,
    const float* __restrict__ bias, unsigned short* __restrict__ C, int relu) {
  int lane = threadIdx.x & 31, wave = threadIdx.x >> 5;
  int wm = wave >> 1, wn = wave & 1;                 // 4x2 waves
  size_t row0 = (size_t)blockIdx.x * 128 + wm * 32;
  int col0 = blockIdx.y * 64 + wn * 32;
  int m = lane & 15;
  int kb = (lane & 16) ? 8 : 0;                      // ISA 16-bit A/B lane K split
  v8f a00 = {}, a01 = {}, a10 = {}, a11 = {};
  const unsigned short* ap = A + (row0 + m) * HPAD + kb;
  const unsigned short* bp = Wt + (size_t)(col0 + m) * HPAD + kb;
  for (int k0 = 0; k0 < HPAD; k0 += 32) {
    Frag fa0, fa1, fb0, fb1;
    fa0.q[0] = *(const uint4*)(ap);              fa0.q[1] = *(const uint4*)(ap + 16);
    fa1.q[0] = *(const uint4*)(ap + 16 * HPAD);  fa1.q[1] = *(const uint4*)(ap + 16 * HPAD + 16);
    fb0.q[0] = *(const uint4*)(bp);              fb0.q[1] = *(const uint4*)(bp + 16);
    fb1.q[0] = *(const uint4*)(bp + 16 * HPAD);  fb1.q[1] = *(const uint4*)(bp + 16 * HPAD + 16);
    a00 = wmma_bf16(fa0.v, fb0.v, a00);
    a01 = wmma_bf16(fa0.v, fb1.v, a01);
    a10 = wmma_bf16(fa1.v, fb0.v, a10);
    a11 = wmma_bf16(fa1.v, fb1.v, a11);
    ap += 32; bp += 32;
    __builtin_prefetch(ap, 0, 1);                // gfx1250 global_prefetch
    __builtin_prefetch(bp, 0, 1);
  }
  store_tiles(C, row0, col0, lane, a00, a01, a10, a11, bias, relu);
}

// ---------------- node projection: h = relu(x[N,64] @ Wn + bn) ----------------
__global__ __launch_bounds__(256) void k_nodeproj(
    const float* __restrict__ X, const unsigned short* __restrict__ Wnt,
    const float* __restrict__ bias, unsigned short* __restrict__ Hout) {
  int lane = threadIdx.x & 31, wave = threadIdx.x >> 5;
  int wm = wave >> 1, wn = wave & 1;
  size_t row0 = (size_t)blockIdx.x * 128 + wm * 32;
  int col0 = blockIdx.y * 64 + wn * 32;
  int m = lane & 15;
  int kb = (lane & 16) ? 8 : 0;
  v8f a00 = {}, a01 = {}, a10 = {}, a11 = {};
  for (int k0 = 0; k0 < 64; k0 += 32) {
    Frag fa0, fa1, fb0, fb1;
    const float* ap0 = X + (row0 + m) * 64 + k0 + kb;
    const float* ap1 = ap0 + (size_t)16 * 64;
#pragma unroll
    for (int h = 0; h < 8; ++h) {
      fa0.u[h]     = f2bf(ap0[h]);
      fa0.u[h + 8] = f2bf(ap0[16 + h]);
      fa1.u[h]     = f2bf(ap1[h]);
      fa1.u[h + 8] = f2bf(ap1[16 + h]);
    }
    const unsigned short* bp = Wnt + (size_t)(col0 + m) * 64 + k0 + kb;
    fb0.q[0] = *(const uint4*)(bp);            fb0.q[1] = *(const uint4*)(bp + 16);
    fb1.q[0] = *(const uint4*)(bp + 16 * 64);  fb1.q[1] = *(const uint4*)(bp + 16 * 64 + 16);
    a00 = wmma_bf16(fa0.v, fb0.v, a00);
    a01 = wmma_bf16(fa0.v, fb1.v, a01);
    a10 = wmma_bf16(fa1.v, fb0.v, a10);
    a11 = wmma_bf16(fa1.v, fb1.v, a11);
  }
  store_tiles(Hout, row0, col0, lane, a00, a01, a10, a11, bias, 1);
}

// ---------------- edge projection: ef = e[E,16] @ We + be (K padded to 32) ----------------
__global__ __launch_bounds__(256) void k_edgeproj(
    const float* __restrict__ Ein, const unsigned short* __restrict__ Wet,
    const float* __restrict__ bias, unsigned short* __restrict__ EF) {
  int lane = threadIdx.x & 31, wave = threadIdx.x >> 5;
  int wm = wave >> 1, wn = wave & 1;
  size_t row0 = (size_t)blockIdx.x * 128 + wm * 32;
  int col0 = blockIdx.y * 64 + wn * 32;
  int m = lane & 15;
  int kb = (lane & 16) ? 8 : 0;
  Frag fa0, fa1, fb0, fb1;
  const float* ap0 = Ein + (row0 + m) * 16 + kb;     // K 0..7 (or 8..15) valid
  const float* ap1 = ap0 + (size_t)16 * 16;
#pragma unroll
  for (int h = 0; h < 8; ++h) {
    fa0.u[h] = f2bf(ap0[h]);  fa0.u[h + 8] = 0;      // K 16..31 is zero pad
    fa1.u[h] = f2bf(ap1[h]);  fa1.u[h + 8] = 0;
  }
  const unsigned short* bp = Wet + (size_t)(col0 + m) * 32 + kb;
  fb0.q[0] = *(const uint4*)(bp);            fb0.q[1] = *(const uint4*)(bp + 16);
  fb1.q[0] = *(const uint4*)(bp + 16 * 32);  fb1.q[1] = *(const uint4*)(bp + 16 * 32 + 16);
  v8f a00 = {}, a01 = {}, a10 = {}, a11 = {};
  a00 = wmma_bf16(fa0.v, fb0.v, a00);
  a01 = wmma_bf16(fa0.v, fb1.v, a01);
  a10 = wmma_bf16(fa1.v, fb0.v, a10);
  a11 = wmma_bf16(fa1.v, fb1.v, a11);
  store_tiles(EF, row0, col0, lane, a00, a01, a10, a11, bias, 0);   // no relu on ef
}

// ---------------- fused message pass: agg[dst] += relu(h[src] + ef) ----------------
__global__ __launch_bounds__(256) void k_message(
    const unsigned short* __restrict__ Hn, const unsigned short* __restrict__ EF,
    const int* __restrict__ src, const int* __restrict__ dst,
    float* __restrict__ agg) {
  int lane = threadIdx.x & 31;
  int e = blockIdx.x * 8 + (threadIdx.x >> 5);       // one edge per wave
  size_t hs = (size_t)src[e] * HPAD;
  size_t es = (size_t)e * HPAD;
  size_t as = (size_t)dst[e] * HPAD;
#pragma unroll
  for (int i = 0; i < 5; ++i) {
    int c = i * 64 + 2 * lane;
    if (c < HREAL) {
      unsigned int hp = *(const unsigned int*)(Hn + hs + c);
      unsigned int ep = *(const unsigned int*)(EF + es + c);
      float m0 = fmaxf(bf2f((unsigned short)hp)          + bf2f((unsigned short)ep),          0.f);
      float m1 = fmaxf(bf2f((unsigned short)(hp >> 16))  + bf2f((unsigned short)(ep >> 16)),  0.f);
      unsafeAtomicAdd(agg + as + c,     m0);          // native global_atomic_add_f32
      unsafeAtomicAdd(agg + as + c + 1, m1);
    }
  }
}

// ---------------- z = h + agg (bf16 out) ----------------
__global__ __launch_bounds__(256) void k_addz(
    const unsigned short* __restrict__ Hn, const float* __restrict__ agg,
    unsigned short* __restrict__ Z) {
  size_t idx = (size_t)blockIdx.x * 256 + threadIdx.x;
  Z[idx] = f2bf(bf2f(Hn[idx]) + agg[idx]);
}

// ---------------- segment-sum pooling: sum[seg[n]] += h[n] ----------------
__global__ __launch_bounds__(256) void k_pool(
    const unsigned short* __restrict__ Hn, const int* __restrict__ seg,
    float* __restrict__ sum) {
  int lane = threadIdx.x & 31;
  int n = blockIdx.x * 8 + (threadIdx.x >> 5);       // one node per wave
  size_t hs = (size_t)n * HPAD;
  size_t os = (size_t)seg[n] * HPAD;
#pragma unroll
  for (int i = 0; i < 5; ++i) {
    int c = i * 64 + 2 * lane;
    if (c < HREAL) {
      unsigned int hp = *(const unsigned int*)(Hn + hs + c);
      unsafeAtomicAdd(sum + os + c,     bf2f((unsigned short)hp));
      unsafeAtomicAdd(sum + os + c + 1, bf2f((unsigned short)(hp >> 16)));
    }
  }
}

// ---------------- final: (r - p, r, p), unpad 320 -> 300 ----------------
__global__ __launch_bounds__(256) void k_final(
    const float* __restrict__ rs, const float* __restrict__ ps, float* __restrict__ out) {
  size_t idx = (size_t)blockIdx.x * 256 + threadIdx.x;   // < NB*300
  size_t b = idx / HREAL;
  size_t c = idx - b * HREAL;
  float r = rs[b * HPAD + c];
  float p = ps[b * HPAD + c];
  out[idx] = r - p;
  out[(size_t)NB * HREAL + idx] = r;
  out[2 * (size_t)NB * HREAL + idx] = p;
}

extern "C" void kernel_launch(void* const* d_in, const int* in_sizes, int n_in,
                              void* d_out, int out_size, void* d_ws, size_t ws_size,
                              hipStream_t stream) {
  (void)in_sizes; (void)n_in; (void)out_size; (void)ws_size;
  const float* r_x = (const float*)d_in[0];
  const float* r_e = (const float*)d_in[1];
  const float* p_x = (const float*)d_in[2];
  const float* p_e = (const float*)d_in[3];
  const float* Wn  = (const float*)d_in[4];
  const float* bn  = (const float*)d_in[5];
  const float* We  = (const float*)d_in[6];
  const float* be  = (const float*)d_in[7];
  const float* W1  = (const float*)d_in[8];
  const float* b1  = (const float*)d_in[9];
  const float* W2  = (const float*)d_in[10];
  const float* b2  = (const float*)d_in[11];
  const int* r_src = (const int*)d_in[12];
  const int* r_dst = (const int*)d_in[13];
  const int* r_seg = (const int*)d_in[14];
  const int* p_src = (const int*)d_in[15];
  const int* p_dst = (const int*)d_in[16];
  const int* p_seg = (const int*)d_in[17];

  char* ws = (char*)d_ws;
  size_t off = 0;
  auto alloc = [&](size_t bytes) -> void* {
    off = (off + 255) & ~(size_t)255;
    void* p = ws + off;
    off += bytes;
    return p;
  };

  unsigned short* Wnt = (unsigned short*)alloc((size_t)HPAD * 64 * 2);
  unsigned short* Wet = (unsigned short*)alloc((size_t)HPAD * 32 * 2);
  unsigned short* W1t = (unsigned short*)alloc((size_t)GDEPTH * HPAD * HPAD * 2);
  unsigned short* W2t = (unsigned short*)alloc((size_t)GDEPTH * HPAD * HPAD * 2);
  float* bnp = (float*)alloc((size_t)HPAD * 4);
  float* bep = (float*)alloc((size_t)HPAD * 4);
  float* b1p = (float*)alloc((size_t)GDEPTH * HPAD * 4);
  float* b2p = (float*)alloc((size_t)GDEPTH * HPAD * 4);
  unsigned short* hbuf  = (unsigned short*)alloc((size_t)NNODES * HPAD * 2);
  unsigned short* efbuf = (unsigned short*)alloc((size_t)NEDGES * HPAD * 2);
  unsigned short* zbuf  = (unsigned short*)alloc((size_t)NNODES * HPAD * 2);
  float* agg  = (float*)alloc((size_t)NNODES * HPAD * 4);
  unsigned short* tbuf = (unsigned short*)agg;   // alias: agg dead after k_addz
  float* rsum = (float*)alloc((size_t)NB * HPAD * 4);
  float* psum = (float*)alloc((size_t)NB * HPAD * 4);

  k_prep<<<(3 * HPAD * HPAD) / 256, 256, 0, stream>>>(
      Wn, bn, We, be, W1, b1, W2, b2, Wnt, Wet, W1t, W2t, bnp, bep, b1p, b2p);
  hipMemsetAsync(rsum, 0, (size_t)NB * HPAD * 4, stream);
  hipMemsetAsync(psum, 0, (size_t)NB * HPAD * 4, stream);

  const float* gx[3]  = { r_x, r_x + (size_t)NNODES * 64, p_x };
  const float* ge[3]  = { r_e, r_e + (size_t)NEDGES * 16, p_e };
  const int* gsrc[3]  = { r_src, r_src + NEDGES, p_src };
  const int* gdst[3]  = { r_dst, r_dst + NEDGES, p_dst };
  const int* gseg[3]  = { r_seg, r_seg + NNODES, p_seg };
  float* gsum[3]      = { rsum, rsum, psum };

  dim3 nodeGrid(NNODES / 128, HPAD / 64);
  dim3 edgeGrid(NEDGES / 128, HPAD / 64);

  for (int g = 0; g < 3; ++g) {
    k_nodeproj<<<nodeGrid, 256, 0, stream>>>(gx[g], Wnt, bnp, hbuf);
    k_edgeproj<<<edgeGrid, 256, 0, stream>>>(ge[g], Wet, bep, efbuf);
    for (int l = 0; l < GDEPTH; ++l) {
      hipMemsetAsync(agg, 0, (size_t)NNODES * HPAD * 4, stream);
      k_message<<<NEDGES / 8, 256, 0, stream>>>(hbuf, efbuf, gsrc[g], gdst[g], agg);
      k_addz<<<((size_t)NNODES * HPAD) / 256, 256, 0, stream>>>(hbuf, agg, zbuf);
      k_gemm320<<<nodeGrid, 256, 0, stream>>>(zbuf, W1t + (size_t)l * HPAD * HPAD,
                                              b1p + l * HPAD, tbuf, 1);
      k_gemm320<<<nodeGrid, 256, 0, stream>>>(tbuf, W2t + (size_t)l * HPAD * HPAD,
                                              b2p + l * HPAD, hbuf, (l < GDEPTH - 1) ? 1 : 0);
    }
    k_pool<<<NNODES / 8, 256, 0, stream>>>(hbuf, gseg[g], gsum[g]);
  }
  k_final<<<(NB * HREAL) / 256, 256, 0, stream>>>(rsum, psum, (float*)d_out);
}
